// UpBlock_25383256720206
// MI455X (gfx1250) — matrix operand order repaired and checked by reading, not compiled, not run on hardware
//
#include <hip/hip_runtime.h>
#include <hip/hip_bf16.h>

typedef __attribute__((ext_vector_type(16))) _Float16 v16h;
typedef __attribute__((ext_vector_type(8)))  _Float16 v8h;
typedef __attribute__((ext_vector_type(8)))  float    v8f;
typedef __attribute__((ext_vector_type(4)))  unsigned int u32x4;
typedef __attribute__((ext_vector_type(8)))  int          i32x8;
typedef __attribute__((ext_vector_type(4)))  int          i32x4;

#define LRELU_SLOPE 0.2f
#define BN_EPS      1e-5f
#define USE_TDM_BUILTIN 1

// ---------------------------------------------------------------------------
// Weight permute + f32->f16.  Reference feature order is f = c*K + k
// (channel-major, neighbor-minor).  We re-order to f' = k*C + c so that a
// 32-wide WMMA K-chunk touches ONE neighbor and 32 contiguous channels.
//   wh[o, k*C + c] = (f16) w[o, c*K + k]
// ---------------------------------------------------------------------------
__global__ void wperm_f16_kernel(const float* __restrict__ w,
                                 _Float16* __restrict__ wh, int C, int K) {
  int idx = blockIdx.x * blockDim.x + threadIdx.x;
  int F = C * K;
  if (idx >= 64 * F) return;
  int o = idx / F;
  int r = idx - o * F;
  int k = r / C;
  int c = r - k * C;
  wh[idx] = (_Float16)w[(size_t)o * F + (size_t)c * K + k];
}

// ---------------------------------------------------------------------------
// up-fc:  hdown[b, n, o] = sum_c x1[b, c, n] * w[o, c] + bias[o]
// 8-way register blocking over o; x1 reads coalesced across n; x1 (42 MB)
// fits in the 192 MB L2 so the 8x re-read per o-group is cheap.
// ---------------------------------------------------------------------------
__global__ __launch_bounds__(256) void upfc_kernel(
    const float* __restrict__ x1, const float* __restrict__ w,
    const float* __restrict__ bias, float* __restrict__ hdown, int Nd) {
  int tid = blockIdx.x * blockDim.x + threadIdx.x;
  if (tid >= 2 * Nd * 8) return;
  int n = tid % Nd;
  int rest = tid / Nd;
  int og = rest & 7;
  int b = rest >> 3;
  float acc[8];
#pragma unroll
  for (int j = 0; j < 8; ++j) acc[j] = bias[og * 8 + j];
  const float* xp = x1 + (size_t)b * 128 * Nd + n;
  const float* wp = w + og * 8 * 128;  // w is (64, 128) row-major
  for (int c = 0; c < 128; ++c) {
    float xv = xp[(size_t)c * Nd];
#pragma unroll
    for (int j = 0; j < 8; ++j) acc[j] = fmaf(xv, wp[j * 128 + c], acc[j]);
  }
  float* op = hdown + ((size_t)b * Nd + n) * 64 + og * 8;
#pragma unroll
  for (int j = 0; j < 8; ++j) op[j] = acc[j];
}

// ---------------------------------------------------------------------------
// Upsample (mean of 2 gathered rows) + concat with x2, written as f16 in
// vertex-major layout  xcat[b, n, c]  (c: 0..63 = upsampled, 64..127 = x2).
// Each thread produces one aligned 16-byte v8h store.
// ---------------------------------------------------------------------------
__global__ __launch_bounds__(256) void upcat_kernel(
    const float* __restrict__ hdown, const float* __restrict__ x2,
    const int* __restrict__ up_neigh, _Float16* __restrict__ xcat,
    int Nd, int Nu) {
  long tid = (long)blockIdx.x * blockDim.x + threadIdx.x;
  long total = (long)2 * 16 * Nu;
  if (tid >= total) return;
  int n = (int)(tid % Nu);
  int rest = (int)(tid / Nu);
  int g = rest & 15;
  int b = rest >> 4;
  v8h outv;
  if (g < 8) {
    int i0 = up_neigh[(size_t)n * 2 + 0];
    int i1 = up_neigh[(size_t)n * 2 + 1];
    const float* p0 = hdown + ((size_t)b * Nd + i0) * 64 + g * 8;
    const float* p1 = hdown + ((size_t)b * Nd + i1) * 64 + g * 8;
#pragma unroll
    for (int j = 0; j < 8; ++j) outv[j] = (_Float16)(0.5f * (p0[j] + p1[j]));
  } else {
    int o0 = (g - 8) * 8;
    const float* q = x2 + ((size_t)b * 64 + o0) * Nu + n;
#pragma unroll
    for (int j = 0; j < 8; ++j) outv[j] = (_Float16)q[(size_t)j * Nu];
  }
  *(v8h*)(xcat + ((size_t)b * Nu + n) * 128 + g * 8) = outv;
}

// ---------------------------------------------------------------------------
// Gathered conv as WMMA GEMM.  One wave32 owns a 16-vertex x 64-channel tile;
// blockIdx.y = batch.  Weights are DMA'd once per workgroup into LDS via the
// Tensor Data Mover (tensor_load_to_lds, TENSORcnt), then all 8 waves read B
// fragments from LDS (ds_load_b128) while global VMEM serves only the
// irregular A gathers.
//   D[m, o] = bias[o] + sum_{k, c}  xin[b, neigh[v_m, k], c] * wh[o, k*C + c]
// A fragment (16x32 f16): lanes 0-15 = rows, K 0..15,
// lanes 16-31 same rows, K 16..31          -> two 16B v8h loads per lane.
// B fragment (32x16 f16): lane = column, 16 contiguous halves (K 0..15 for
// lanes<16, 16..31 for lanes>=16)          -> two 16B v8h loads per lane.
// EXEC stays all-ones through every WMMA (tail rows clamp loads, guard stores).
// ---------------------------------------------------------------------------
template <int C>
__global__ __launch_bounds__(256) void ico_conv_wmma_kernel(
    const _Float16* __restrict__ xin,   // (2, N, C)   f16
    const int* __restrict__ neigh,      // (N, 7)
    const _Float16* __restrict__ wh,    // (64, 7*C)   f16, permuted
    const float* __restrict__ bias,     // (64,)
    float* __restrict__ yout,           // (2, N, 64)  f32
    int N) {
  constexpr int K = 7;
  constexpr int F = K * C;           // 896 (conv1) / 448 (conv2)
  constexpr int CH32 = C / 32;
  constexpr unsigned WBYTES = 64u * F * 2u;  // 114688 / 57344 bytes
  extern __shared__ _Float16 lds_w[];

  const int lane = threadIdx.x & 31;
  const int wave = threadIdx.x >> 5;

  // ---- stage the full weight matrix into LDS (once per workgroup) ----
#if USE_TDM_BUILTIN && __has_builtin(__builtin_amdgcn_tensor_load_to_lds)
  if (wave == 0) {
    // Tensor DMA Descriptor (CDNA5 ISA):
    // 1-D tile, data_size=8B, nelem = WBYTES/8 elements, no padding/multicast.
    const unsigned nelem = WBYTES / 8u;
    const unsigned lds_off = (unsigned)(size_t)lds_w;        // LDS byte offset
    unsigned long long ga = (unsigned long long)(size_t)wh;  // global byte addr
    u32x4 g0;
    g0[0] = 1u;                                          // count=1 (valid)
    g0[1] = lds_off;                                     // lds_addr
    g0[2] = (unsigned)(ga & 0xFFFFFFFFu);                // global_addr[31:0]
    g0[3] = (unsigned)((ga >> 32) & 0x01FFFFFFu) | (2u << 30);  // addr hi, type=2
    i32x8 g1;
    g1[0] = (int)(3u << 16);                             // data_size = 8 bytes
    g1[1] = (int)((nelem & 0xFFFFu) << 16);              // tensor_dim0[15:0]
    g1[2] = (int)(((nelem >> 16) & 0xFFFFu) | (1u << 16));  // dim0 hi, tensor_dim1=1
    g1[3] = (int)(nelem << 16);                          // tile_dim0 = nelem
    g1[4] = 0;                                           // tile_dim1/2 unused
    g1[5] = (int)nelem;                                  // tensor_dim0_stride lo
    g1[6] = 0;
    g1[7] = 0;
    i32x4 z4 = {0, 0, 0, 0};
#if __clang_major__ >= 23
    i32x8 z8 = {0, 0, 0, 0, 0, 0, 0, 0};
    __builtin_amdgcn_tensor_load_to_lds(g0, g1, z4, z4, z8, 0);
#else
    __builtin_amdgcn_tensor_load_to_lds(g0, g1, z4, z4, 0);
#endif
    __builtin_amdgcn_s_wait_tensorcnt(0);
  }
  __syncthreads();
#else
  for (unsigned i = threadIdx.x; i < WBYTES / 16u; i += blockDim.x)
    ((v8h*)lds_w)[i] = ((const v8h*)wh)[i];
  __syncthreads();
#endif

  const int tilesPerBatch = (N + 15) >> 4;
  const int tile = blockIdx.x * (blockDim.x >> 5) + wave;
  if (tile >= tilesPerBatch) return;  // wave-uniform exit (after barrier)
  const int b = blockIdx.y;
  const int row = lane & 15;
  const int hi = lane >> 4;

  int v_row = tile * 16 + row;
  if (v_row >= N) v_row = N - 1;  // clamp: keep EXEC full for WMMA

  v8f acc[4];
#pragma unroll
  for (int nt = 0; nt < 4; ++nt) {
    float bv = bias[nt * 16 + row];  // C/D layout: N-column = lane & 15
    acc[nt] = (v8f){bv, bv, bv, bv, bv, bv, bv, bv};
  }

  const size_t xbase = (size_t)b * N * C;
  const _Float16* wlane = lds_w + (size_t)row * F + hi * 16;

  for (int k = 0; k < K; ++k) {
    const int nb = neigh[(size_t)v_row * K + k];
    const _Float16* arow = xin + xbase + (size_t)nb * C + hi * 16;
#pragma unroll
    for (int cc = 0; cc < CH32; ++cc) {
      const int f0 = (k * CH32 + cc) * 32;
      v8h alo = *(const v8h*)(arow + cc * 32);
      v8h ahi = *(const v8h*)(arow + cc * 32 + 8);
      v16h afrag = __builtin_shufflevector(
          alo, ahi, 0, 1, 2, 3, 4, 5, 6, 7, 8, 9, 10, 11, 12, 13, 14, 15);
#pragma unroll
      for (int nt = 0; nt < 4; ++nt) {
        const _Float16* bp = wlane + (size_t)nt * 16 * F + f0;
        v8h blo = *(const v8h*)bp;        // ds_load_b128
        v8h bhi = *(const v8h*)(bp + 8);  // ds_load_b128
        v16h bfrag = __builtin_shufflevector(
            blo, bhi, 0, 1, 2, 3, 4, 5, 6, 7, 8, 9, 10, 11, 12, 13, 14, 15);
        acc[nt] = __builtin_amdgcn_wmma_f32_16x16x32_f16(
            false, afrag, false, bfrag, (short)0, acc[nt], false, false);
      }
    }
  }

  // D layout: VGPR r of lanes<16 -> M=r; lanes>=16 -> M=r+8; column = row
#pragma unroll
  for (int nt = 0; nt < 4; ++nt) {
#pragma unroll
    for (int r = 0; r < 8; ++r) {
      int vm = tile * 16 + r + hi * 8;
      if (vm < N)
        yout[((size_t)b * N + vm) * 64 + nt * 16 + row] = acc[nt][r];
    }
  }
}

// ---------------------------------------------------------------------------
// BatchNorm helpers: stats[0..63] = sum, stats[64..127] = sum of squares
// ---------------------------------------------------------------------------
__global__ void bn_zero_kernel(float* __restrict__ stats) {
  if (blockIdx.x == 0 && threadIdx.x < 128) stats[threadIdx.x] = 0.0f;
}

__global__ __launch_bounds__(256) void bn_reduce_kernel(
    const float* __restrict__ y, float* __restrict__ stats, long rows) {
  __shared__ float ssum[256];
  __shared__ float ssq[256];
  int ch = threadIdx.x & 63;
  int sub = threadIdx.x >> 6;  // 0..3
  float s = 0.0f, q = 0.0f;
  long stride = (long)gridDim.x * 4;
  for (long r = (long)blockIdx.x * 4 + sub; r < rows; r += stride) {
    float v = y[r * 64 + ch];
    s += v;
    q += v * v;
  }
  ssum[threadIdx.x] = s;
  ssq[threadIdx.x] = q;
  __syncthreads();
  if (sub == 0) {
    s = ssum[ch] + ssum[64 + ch] + ssum[128 + ch] + ssum[192 + ch];
    q = ssq[ch] + ssq[64 + ch] + ssq[128 + ch] + ssq[192 + ch];
    atomicAdd(&stats[ch], s);
    atomicAdd(&stats[64 + ch], q);
  }
}

// BN + LeakyReLU, emit f16 (B, N, 64) as next conv's A operand
__global__ __launch_bounds__(256) void bn_act_h_kernel(
    const float* __restrict__ y, const float* __restrict__ stats,
    const float* __restrict__ gamma, const float* __restrict__ beta,
    _Float16* __restrict__ xn, long rows) {
  long tid = (long)blockIdx.x * blockDim.x + threadIdx.x;
  if (tid >= rows * 64) return;
  int ch = (int)(tid & 63);
  float invN = 1.0f / (float)rows;
  float m = stats[ch] * invN;
  float var = stats[64 + ch] * invN - m * m;
  float sc = gamma[ch] * rsqrtf(var + BN_EPS);
  float sh = beta[ch] - m * sc;
  float v = fmaf(y[tid], sc, sh);
  v = (v >= 0.0f) ? v : LRELU_SLOPE * v;
  xn[tid] = (_Float16)v;
}

// Final BN + LeakyReLU with transpose to reference layout (B, 64, N_UP) f32.
// Coalesced stores; strided reads land in L2 (y2 ~84 MB vs 192 MB L2).
__global__ __launch_bounds__(256) void bn_act_out_kernel(
    const float* __restrict__ y, const float* __restrict__ stats,
    const float* __restrict__ gamma, const float* __restrict__ beta,
    float* __restrict__ out, int Nu) {
  long tid = (long)blockIdx.x * blockDim.x + threadIdx.x;
  long total = (long)2 * 64 * Nu;
  if (tid >= total) return;
  int n = (int)(tid % Nu);
  int rest = (int)(tid / Nu);
  int ch = rest & 63;
  int b = rest >> 6;
  float invN = 1.0f / (float)((long)2 * Nu);
  float m = stats[ch] * invN;
  float var = stats[64 + ch] * invN - m * m;
  float sc = gamma[ch] * rsqrtf(var + BN_EPS);
  float sh = beta[ch] - m * sc;
  float v = fmaf(y[((size_t)b * Nu + n) * 64 + ch], sc, sh);
  out[tid] = (v >= 0.0f) ? v : LRELU_SLOPE * v;
}

// ---------------------------------------------------------------------------
extern "C" void kernel_launch(void* const* d_in, const int* in_sizes, int n_in,
                              void* d_out, int out_size, void* d_ws,
                              size_t ws_size, hipStream_t stream) {
  const float* x1       = (const float*)d_in[0];   // (2, 128, Nd)
  const float* x2       = (const float*)d_in[1];   // (2, 64, Nu)
  const float* up_fc_w  = (const float*)d_in[2];   // (64, 128)
  const float* up_fc_b  = (const float*)d_in[3];   // (64,)
  const float* conv1_w  = (const float*)d_in[4];   // (64, 896)
  const float* conv1_b  = (const float*)d_in[5];   // (64,)
  const float* bn1_g    = (const float*)d_in[6];
  const float* bn1_b    = (const float*)d_in[7];
  const float* conv2_w  = (const float*)d_in[8];   // (64, 448)
  const float* conv2_b  = (const float*)d_in[9];
  const float* bn2_g    = (const float*)d_in[10];
  const float* bn2_b    = (const float*)d_in[11];
  const int* up_neigh   = (const int*)d_in[12];    // (Nu, 2)
  const int* conv_neigh = (const int*)d_in[13];    // (Nu, 7)

  const int Nd = in_sizes[0] / (2 * 128);
  const int Nu = in_sizes[13] / 7;

  // ---- workspace carve-out (256B aligned regions) ----
  auto a256 = [](size_t x) { return (x + 255) & ~(size_t)255; };
  char* p = (char*)d_ws;
  float*    hdown = (float*)p;    p += a256((size_t)2 * Nd * 64 * 4);
  _Float16* xcat  = (_Float16*)p; p += a256((size_t)2 * Nu * 128 * 2);
  _Float16* w1h   = (_Float16*)p; p += a256((size_t)64 * 896 * 2);
  _Float16* w2h   = (_Float16*)p; p += a256((size_t)64 * 448 * 2);
  float*    ybuf  = (float*)p;    p += a256((size_t)2 * Nu * 64 * 4);
  _Float16* xn    = (_Float16*)p; p += a256((size_t)2 * Nu * 64 * 2);
  float*    stats = (float*)p;    p += a256((size_t)128 * 4);

  const int T = 256;
  auto blocks = [&](long total) { return (int)((total + T - 1) / T); };

  // 1) weight permute + convert
  wperm_f16_kernel<<<blocks(64L * 896), T, 0, stream>>>(conv1_w, w1h, 128, 7);
  wperm_f16_kernel<<<blocks(64L * 448), T, 0, stream>>>(conv2_w, w2h, 64, 7);

  // 2) up-fc GEMM on down-sphere, then gather-mean upsample + concat (f16)
  upfc_kernel<<<blocks((long)2 * Nd * 8), T, 0, stream>>>(x1, up_fc_w, up_fc_b,
                                                          hdown, Nd);
  upcat_kernel<<<blocks((long)2 * 16 * Nu), T, 0, stream>>>(hdown, x2, up_neigh,
                                                            xcat, Nd, Nu);

  // 3) conv1 (WMMA + TDM-staged weights), BN1 stats, BN1+LReLU -> f16
  const int tilesPB = (Nu + 15) / 16;
  const dim3 convGrid((tilesPB + 7) / 8, 2);  // 8 waves/block, y = batch
  ico_conv_wmma_kernel<128><<<convGrid, T, 64 * 896 * 2, stream>>>(
      xcat, conv_neigh, w1h, conv1_b, ybuf, Nu);
  bn_zero_kernel<<<1, 128, 0, stream>>>(stats);
  bn_reduce_kernel<<<512, T, 0, stream>>>(ybuf, stats, (long)2 * Nu);
  bn_act_h_kernel<<<blocks((long)2 * Nu * 64), T, 0, stream>>>(
      ybuf, stats, bn1_g, bn1_b, xn, (long)2 * Nu);

  // 4) conv2 (WMMA + TDM-staged weights), BN2 stats, BN2+LReLU -> output
  ico_conv_wmma_kernel<64><<<convGrid, T, 64 * 448 * 2, stream>>>(
      xn, conv_neigh, w2h, conv2_b, ybuf, Nu);
  bn_zero_kernel<<<1, 128, 0, stream>>>(stats);
  bn_reduce_kernel<<<512, T, 0, stream>>>(ybuf, stats, (long)2 * Nu);
  bn_act_out_kernel<<<blocks((long)2 * 64 * Nu), T, 0, stream>>>(
      ybuf, stats, bn2_g, bn2_b, (float*)d_out, Nu);
}
